// TransformerLayer_13743895347377
// MI455X (gfx1250) — matrix operand order repaired and checked by reading, not compiled, not run on hardware
//
#include <hip/hip_runtime.h>
#include <hip/hip_bf16.h>
#include <math.h>

// ---------------------------------------------------------------------------
// Transformer layer for MI455X (gfx1250):
//  - all GEMMs via v_wmma_f32_16x16x32_f16 (f32 accumulate)
//  - GEMM A/B panels staged global->LDS with GLOBAL_LOAD_ASYNC_TO_LDS_B128,
//    double-buffered, synchronized with s_wait_asynccnt
//  - flash-style causal attention, online softmax via half-wave shfl_xor
// ---------------------------------------------------------------------------

typedef __attribute__((ext_vector_type(16))) _Float16 v16h;
typedef __attribute__((ext_vector_type(8)))  float    v8f;
typedef __attribute__((ext_vector_type(4)))  int      v4i;

#define Dm   1024
#define Hh   16
#define DH   64
#define Ff   4096
#define Bb   16
#define Ss   512
#define TOK  (Bb * Ss)              // 8192 rows
#define QELEMS (Bb * Hh * Ss * DH)  // 8,388,608 f16 elems per Q/K/V buffer

// ---- WMMA wrapper ----------------------------------------------------------
__device__ inline v8f wmma16(v16h a, v16h b, v8f c) {
  return __builtin_amdgcn_wmma_f32_16x16x32_f16(
      /*neg_a=*/false, a, /*neg_b=*/false, b,
      /*c_mod=*/(short)0, c, /*reuse_a=*/false, /*reuse_b=*/false);
}

// ---- Async global->LDS copy (CDNA5, ASYNCcnt) ------------------------------
__device__ inline void async_b128(const _Float16* g, _Float16* l) {
#if __has_builtin(__builtin_amdgcn_global_load_async_to_lds_b128)
  _Float16* gnc = const_cast<_Float16*>(g);
  __builtin_amdgcn_global_load_async_to_lds_b128(
      (__attribute__((address_space(1))) v4i*)gnc,
      (__attribute__((address_space(3))) v4i*)l, 0, 0);
#else
  unsigned loff = (unsigned)(__SIZE_TYPE__)(__attribute__((address_space(3))) _Float16*)l;
  asm volatile("global_load_async_to_lds_b128 %0, %1, off"
               :: "v"(loff), "v"(g) : "memory");
#endif
}

template <int N>
__device__ inline void wait_async() {
#if __has_builtin(__builtin_amdgcn_s_wait_asynccnt)
  __builtin_amdgcn_s_wait_asynccnt(N);
#else
  asm volatile("s_wait_asynccnt %0" :: "i"(N) : "memory");
#endif
}

// ---- Tile loaders, contiguous k order shared by A and B (work for global or LDS) -----
// A (16x32 f16): lane = row (lane&15). lanes 0-15 hold K 0..15,
// lanes 16-31 hold K 16..31 (contiguous).
__device__ inline v16h ld_a_tile(const _Float16* base, int ld, int row, int k, int lane) {
  const _Float16* p = base + (size_t)(row + (lane & 15)) * ld + k + ((lane >> 4) << 4);
  union { v16h v; uint4 q[2]; } u;
  u.q[0] = *(const uint4*)(p);
  u.q[1] = *(const uint4*)(p + 8);
  return u.v;
}
// B (32x16 f16), sourced from Bt[N][K] (row-major over K): lane = column
// (lane&15). lanes 0-15 hold K 0..15, lanes 16-31 hold K 16..31 (contiguous).
__device__ inline v16h ld_b_tile(const _Float16* base, int ld, int col, int k, int lane) {
  const _Float16* p = base + (size_t)(col + (lane & 15)) * ld + k + ((lane >> 4) << 4);
  union { v16h v; uint4 q[2]; } u;
  u.q[0] = *(const uint4*)(p);
  u.q[1] = *(const uint4*)(p + 8);
  return u.v;
}

// ---- Weight conversion / transposition kernels -----------------------------
__global__ __launch_bounds__(256) void cvt_qkvT(const float* __restrict__ Wq,
                                                const float* __restrict__ Wk,
                                                const float* __restrict__ Wv,
                                                _Float16* __restrict__ T) {
  int i = blockIdx.x * 256 + threadIdx.x;
  if (i >= 3 * 1024 * 1024) return;
  int which = i >> 20;
  int rem   = i & 1048575;
  int n = rem >> 10, d = rem & 1023;
  int h = n >> 6, e = n & 63;
  const float* W = (which == 0) ? Wq : (which == 1) ? Wk : Wv;
  T[i] = (_Float16)W[((size_t)h * 1024 + d) * 64 + e];
}

// T[n*rows + d] = W[d*cols + n]   (W is [rows][cols] f32, T is [cols][rows] f16)
__global__ __launch_bounds__(256) void cvt_transpose(const float* __restrict__ W,
                                                     _Float16* __restrict__ T,
                                                     int rows, int cols) {
  int i = blockIdx.x * 256 + threadIdx.x;
  if (i >= rows * cols) return;
  int n = i / rows, d = i % rows;
  T[i] = (_Float16)W[(size_t)d * cols + n];
}

// ---- LayerNorm (f32 in -> f16 out), one block per row of 1024 --------------
__global__ __launch_bounds__(256) void ln_kernel(const float* __restrict__ X,
                                                 const float* __restrict__ gamma,
                                                 const float* __restrict__ beta,
                                                 _Float16* __restrict__ out) {
  int row = blockIdx.x;
  const float* xr = X + (size_t)row * Dm;
  float s = 0.f, s2 = 0.f;
  for (int i = threadIdx.x; i < Dm; i += 256) {
    float v = xr[i]; s += v; s2 += v * v;
  }
  for (int o = 16; o > 0; o >>= 1) {
    s  += __shfl_xor(s,  o, 32);
    s2 += __shfl_xor(s2, o, 32);
  }
  __shared__ float rs[8], rs2[8];
  int wid = threadIdx.x >> 5, lane = threadIdx.x & 31;
  if (lane == 0) { rs[wid] = s; rs2[wid] = s2; }
  __syncthreads();
  if (wid == 0) {
    s  = (lane < 8) ? rs[lane]  : 0.f;
    s2 = (lane < 8) ? rs2[lane] : 0.f;
    for (int o = 4; o > 0; o >>= 1) {
      s  += __shfl_xor(s,  o, 32);
      s2 += __shfl_xor(s2, o, 32);
    }
    if (lane == 0) { rs[0] = s; rs2[0] = s2; }
  }
  __syncthreads();
  float mean = rs[0] * (1.f / Dm);
  float var  = rs2[0] * (1.f / Dm) - mean * mean;
  float rstd = rsqrtf(var + 1e-5f);
  for (int i = threadIdx.x; i < Dm; i += 256)
    out[(size_t)row * Dm + i] = (_Float16)(gamma[i] * (xr[i] - mean) * rstd + beta[i]);
}

// ---- WMMA GEMM with async-LDS double buffering -----------------------------
// C[M,N] = A[M,K] x Bt[N,K]^T.  Block = 256 threads = 8 waves (2x4) owning a
// 64(M) x 128(N) tile; each wave computes 32x32 via 4 accumulators.
// Per k-step (32): stage A panel 64x32 (4 KB) + B panel 128x32 (8 KB) into
// LDS with GLOBAL_LOAD_ASYNC_TO_LDS_B128, double buffered.
// MODE 0: QKV projection (outH = base of Q|K|Vt, special scatter, Q *= 1/8)
// MODE 1: outF[m,n] = acc + bias[n] + resid[m,n]        (f32 out)
// MODE 2: outH[m,n] = gelu(acc + bias[n])               (f16 out, exact erf)
template <int MODE>
__global__ __launch_bounds__(256) void gemm_wmma(const _Float16* __restrict__ A,
                                                 const _Float16* __restrict__ Bt,
                                                 int K,
                                                 const float* __restrict__ bias,
                                                 const float* __restrict__ resid,
                                                 float* __restrict__ outF,
                                                 _Float16* __restrict__ outH,
                                                 int ldo) {
  __shared__ _Float16 lA[2][64 * 32];    //  8 KB
  __shared__ _Float16 lB[2][128 * 32];   // 16 KB
  int t    = threadIdx.x;
  int lane = t & 31;
  int wid  = t >> 5;
  int m0B = blockIdx.y * 64;
  int n0B = blockIdx.x * 128;
  int mloc = (wid >> 2) * 32;
  int nloc = (wid & 3) * 32;

  // staging decomposition: one b128 (8 halves) per (row, chunk)
  int arow = t >> 2, achk = (t & 3) * 8;                 // A: 64 rows x 4 chunks
  int b0row = t >> 2, b0chk = (t & 3) * 8;               // B: 128 rows x 4 chunks
  int b1row = (t + 256) >> 2, b1chk = ((t + 256) & 3) * 8;

  auto stage = [&](int buf, int k) {
    async_b128(A  + (size_t)(m0B + arow)  * K + k + achk,  &lA[buf][arow * 32 + achk]);
    async_b128(Bt + (size_t)(n0B + b0row) * K + k + b0chk, &lB[buf][b0row * 32 + b0chk]);
    async_b128(Bt + (size_t)(n0B + b1row) * K + k + b1chk, &lB[buf][b1row * 32 + b1chk]);
  };

  v8f acc[2][2] = {};
  stage(0, 0);
  int cur = 0;
  for (int k = 0; k < K; k += 32) {
    int knext = (k + 32 < K) ? (k + 32) : 0;   // harmless reload on last iter
    stage(cur ^ 1, knext);
    wait_async<3>();      // in-order completion => batch `cur` has landed
    __syncthreads();      // ... for every wave in the block

    v16h a0 = ld_a_tile(&lA[cur][0], 32, mloc,      0, lane);
    v16h a1 = ld_a_tile(&lA[cur][0], 32, mloc + 16, 0, lane);
    v16h b0 = ld_b_tile(&lB[cur][0], 32, nloc,      0, lane);
    v16h b1 = ld_b_tile(&lB[cur][0], 32, nloc + 16, 0, lane);
    acc[0][0] = wmma16(a0, b0, acc[0][0]);
    acc[0][1] = wmma16(a0, b1, acc[0][1]);
    acc[1][0] = wmma16(a1, b0, acc[1][0]);
    acc[1][1] = wmma16(a1, b1, acc[1][1]);

    __syncthreads();      // everyone done reading buf `cur` before overwrite
    cur ^= 1;
  }

  int laneHalf = lane >> 4, col = lane & 15;
  for (int tm = 0; tm < 2; tm++)
    for (int tn = 0; tn < 2; tn++)
      for (int v = 0; v < 8; v++) {
        int m = m0B + mloc + tm * 16 + v + laneHalf * 8;
        int n = n0B + nloc + tn * 16 + col;
        float val = acc[tm][tn][v];
        if constexpr (MODE == 0) {
          int bI = m >> 9, sI = m & 511;
          if (n < 1024) {               // Q, pre-scaled by 1/sqrt(DH)
            int h = n >> 6, e = n & 63;
            outH[((size_t)(bI * Hh + h) * Ss + sI) * DH + e] = (_Float16)(val * 0.125f);
          } else if (n < 2048) {        // K
            int n2 = n - 1024, h = n2 >> 6, e = n2 & 63;
            outH[QELEMS + ((size_t)(bI * Hh + h) * Ss + sI) * DH + e] = (_Float16)val;
          } else {                      // V stored transposed: [B,H,DH,S]
            int n2 = n - 2048, h = n2 >> 6, e = n2 & 63;
            outH[2 * (size_t)QELEMS + ((size_t)(bI * Hh + h) * DH + e) * Ss + sI] =
                (_Float16)val;
          }
        } else if constexpr (MODE == 1) {
          float r = val + bias[n] + resid[(size_t)m * ldo + n];
          outF[(size_t)m * ldo + n] = r;
        } else {  // MODE == 2 : bias + exact GELU -> f16
          float z = val + bias[n];
          float g = 0.5f * z * (1.0f + erff(z * 0.70710678118654752f));
          outH[(size_t)m * ldo + n] = (_Float16)g;
        }
      }
}

// ---- Causal flash attention: one wave per 16 query rows --------------------
__global__ __launch_bounds__(128) void attn_kernel(const _Float16* __restrict__ Q,
                                                   const _Float16* __restrict__ Kb,
                                                   const _Float16* __restrict__ Vt,
                                                   _Float16* __restrict__ concat) {
  __shared__ _Float16 plds[4][16 * 32];  // per-wave P tile (16 q x 32 k)
  int lane = threadIdx.x & 31;
  int w = blockIdx.x * 4 + (threadIdx.x >> 5);
  int qt = w & 31, h = (w >> 5) & 15, b = w >> 9;
  int q0 = qt * 16;
  int laneHalf = lane >> 4, col = lane & 15;

  const _Float16* Qh = Q  + (size_t)(b * Hh + h) * Ss * DH;
  const _Float16* Kh = Kb + (size_t)(b * Hh + h) * Ss * DH;
  const _Float16* Vh = Vt + (size_t)(b * Hh + h) * DH * Ss;
  _Float16* pl = plds[threadIdx.x >> 5];

  v16h qlo = ld_a_tile(Qh, DH, q0, 0, lane);   // dh 0..31
  v16h qhi = ld_a_tile(Qh, DH, q0, 32, lane);  // dh 32..63

  v8f o[4] = {};
  float mi[8], li[8];
  for (int v = 0; v < 8; v++) { mi[v] = -1e30f; li[v] = 0.f; }

  for (int kb = 0; kb < q0 + 16; kb += 32) {
    // scores: S = Q x K^T, contraction over dh (2 WMMAs per 16-key tile)
    v16h k0lo = ld_b_tile(Kh, DH, kb,      0,  lane);
    v16h k0hi = ld_b_tile(Kh, DH, kb,      32, lane);
    v16h k1lo = ld_b_tile(Kh, DH, kb + 16, 0,  lane);
    v16h k1hi = ld_b_tile(Kh, DH, kb + 16, 32, lane);
    v8f s0 = {}, s1 = {};
    s0 = wmma16(qlo, k0lo, s0); s0 = wmma16(qhi, k0hi, s0);
    s1 = wmma16(qlo, k1lo, s1); s1 = wmma16(qhi, k1hi, s1);

    float p0[8], p1[8];
    for (int v = 0; v < 8; v++) {
      int m = v + laneHalf * 8;
      int qg = q0 + m;
      float x0 = (kb + col      <= qg) ? s0[v] : -1e30f;  // causal mask
      float x1 = (kb + 16 + col <= qg) ? s1[v] : -1e30f;
      float r = fmaxf(x0, x1);
      r = fmaxf(r, __shfl_xor(r, 1, 32));
      r = fmaxf(r, __shfl_xor(r, 2, 32));
      r = fmaxf(r, __shfl_xor(r, 4, 32));
      r = fmaxf(r, __shfl_xor(r, 8, 32));
      float nm = fmaxf(mi[v], r);
      float sc = __expf(mi[v] - nm);
      float e0 = __expf(x0 - nm), e1 = __expf(x1 - nm);
      float rs = e0 + e1;
      rs += __shfl_xor(rs, 1, 32);
      rs += __shfl_xor(rs, 2, 32);
      rs += __shfl_xor(rs, 4, 32);
      rs += __shfl_xor(rs, 8, 32);
      li[v] = li[v] * sc + rs;
      mi[v] = nm;
      p0[v] = e0; p1[v] = e1;
      for (int t = 0; t < 4; t++) o[t][v] *= sc;  // rescale running output
    }

    // Re-shape P from C-layout to A-layout through LDS (wave-private tile).
    for (int v = 0; v < 8; v++) {
      int m = v + laneHalf * 8;
      pl[m * 32 + col]      = (_Float16)p0[v];
      pl[m * 32 + 16 + col] = (_Float16)p1[v];
    }
    v16h pa = ld_a_tile(pl, 32, 0, 0, lane);

    // O += P x V, contraction over 32 keys; Vt gives contiguous B loads.
    for (int t = 0; t < 4; t++) {
      v16h vb = ld_b_tile(Vh, Ss, t * 16, kb, lane);
      o[t] = wmma16(pa, vb, o[t]);
    }
  }

  // Normalize and write concat[b][s][h*64 + dh]  (f16 input for Wo GEMM)
  for (int t = 0; t < 4; t++)
    for (int v = 0; v < 8; v++) {
      int m = v + laneHalf * 8;
      float val = o[t][v] / li[v];
      concat[((size_t)(b * Ss) + q0 + m) * Dm + h * DH + t * 16 + col] = (_Float16)val;
    }
}

// ---------------------------------------------------------------------------
extern "C" void kernel_launch(void* const* d_in, const int* in_sizes, int n_in,
                              void* d_out, int out_size, void* d_ws, size_t ws_size,
                              hipStream_t stream) {
  const float* x      = (const float*)d_in[0];
  const float* Wq     = (const float*)d_in[1];
  const float* Wk     = (const float*)d_in[2];
  const float* Wv     = (const float*)d_in[3];
  const float* Wo     = (const float*)d_in[4];
  const float* bo     = (const float*)d_in[5];
  const float* gamma1 = (const float*)d_in[6];
  const float* beta1  = (const float*)d_in[7];
  const float* gamma2 = (const float*)d_in[8];
  const float* beta2  = (const float*)d_in[9];
  const float* W1     = (const float*)d_in[10];
  const float* b1     = (const float*)d_in[11];
  const float* W2     = (const float*)d_in[12];
  const float* b2     = (const float*)d_in[13];
  float* out = (float*)d_out;

  // Carve workspace (256B aligned chunks).
  char* p = (char*)d_ws;
  auto carve = [&](size_t bytes) {
    void* r = (void*)p;
    p += (bytes + 255) & ~(size_t)255;
    return r;
  };
  _Float16* n1     = (_Float16*)carve((size_t)TOK * Dm * 2);          // 16.8 MB
  _Float16* qkvT   = (_Float16*)carve((size_t)3 * 1024 * 1024 * 2);   //  6.3 MB
  _Float16* woT    = (_Float16*)carve((size_t)1024 * 1024 * 2);       //  2.1 MB
  _Float16* w1T    = (_Float16*)carve((size_t)4096 * 1024 * 2);       //  8.4 MB
  _Float16* w2T    = (_Float16*)carve((size_t)1024 * 4096 * 2);       //  8.4 MB
  _Float16* qkvbuf = (_Float16*)carve((size_t)3 * QELEMS * 2);        // 50.3 MB (Q|K|Vt)
  _Float16* concat = (_Float16*)carve((size_t)TOK * Dm * 2);          // 16.8 MB
  float*    x2     = (float*)   carve((size_t)TOK * Dm * 4);          // 33.6 MB
  _Float16* n2     = (_Float16*)carve((size_t)TOK * Dm * 2);          // 16.8 MB
  _Float16* hbuf   = (_Float16*)carve((size_t)TOK * Ff * 2);          // 67.1 MB

  _Float16* Qb  = qkvbuf;
  _Float16* Kbf = qkvbuf + QELEMS;
  _Float16* Vtb = qkvbuf + 2 * (size_t)QELEMS;

  // 1) weight conversion (f32 -> f16, transposed to [N][K])
  cvt_qkvT<<<12288, 256, 0, stream>>>(Wq, Wk, Wv, qkvT);
  cvt_transpose<<<4096, 256, 0, stream>>>(Wo, woT, 1024, 1024);
  cvt_transpose<<<16384, 256, 0, stream>>>(W1, w1T, 1024, 4096);
  cvt_transpose<<<16384, 256, 0, stream>>>(W2, w2T, 4096, 1024);

  // 2) LN1
  ln_kernel<<<TOK, 256, 0, stream>>>(x, gamma1, beta1, n1);

  // 3) fused QKV projection: [8192,1024] x [1024,3072]
  gemm_wmma<0><<<dim3(3072 / 128, TOK / 64), 256, 0, stream>>>(
      n1, qkvT, Dm, nullptr, nullptr, nullptr, qkvbuf, 0);

  // 4) causal attention (8192 wave-tasks, 4 waves / block)
  attn_kernel<<<2048, 128, 0, stream>>>(Qb, Kbf, Vtb, concat);

  // 5) output projection + bias + residual -> x2 (f32)
  gemm_wmma<1><<<dim3(1024 / 128, TOK / 64), 256, 0, stream>>>(
      concat, woT, Dm, bo, x, x2, nullptr, Dm);

  // 6) LN2
  ln_kernel<<<TOK, 256, 0, stream>>>(x2, gamma2, beta2, n2);

  // 7) FFN up + exact GELU -> hbuf (f16)
  gemm_wmma<2><<<dim3(Ff / 128, TOK / 64), 256, 0, stream>>>(
      n2, w1T, Dm, b1, nullptr, nullptr, hbuf, Ff);

  // 8) FFN down + bias + residual -> d_out (f32)
  gemm_wmma<1><<<dim3(1024 / 128, TOK / 64), 256, 0, stream>>>(
      hbuf, w2T, Ff, b2, x2, out, nullptr, Dm);
}
